// GaussianNet_60979945668830
// MI455X (gfx1250) — matrix operand order repaired and checked by reading, not compiled, not run on hardware
//
#include <hip/hip_runtime.h>
#include <hip/hip_bf16.h>
#include <math.h>

// ---------------- problem constants ----------------
#define NV    27554              // vertices
#define NP    27648              // NV padded to 128 (216*128) for GEMM M
#define TCLIP 8
#define NJ    24
#define INDIM 2342               // 8*291 + 14
#define KP    2368               // INDIM padded to 32 (74*32) for GEMM K
#define HID   512

typedef __attribute__((ext_vector_type(16))) _Float16 v16h;
typedef __attribute__((ext_vector_type(8)))  float    v8f;

// =====================================================================
// 1) joints = J_regressor @ v_template   (24 x 3)
// =====================================================================
__global__ __launch_bounds__(256) void k_joints(const float* __restrict__ Jr,
                                                const float* __restrict__ vt,
                                                float* __restrict__ joints) {
  __shared__ float red[256 * 3];
  const int j = blockIdx.x;
  float sx = 0.f, sy = 0.f, sz = 0.f;
  for (int n = threadIdx.x; n < NV; n += 256) {
    float w = Jr[(size_t)j * NV + n];
    sx += w * vt[n * 3 + 0];
    sy += w * vt[n * 3 + 1];
    sz += w * vt[n * 3 + 2];
  }
  red[threadIdx.x]       = sx;
  red[256 + threadIdx.x] = sy;
  red[512 + threadIdx.x] = sz;
  __syncthreads();
  for (int s = 128; s > 0; s >>= 1) {
    if (threadIdx.x < (unsigned)s) {
      red[threadIdx.x]       += red[threadIdx.x + s];
      red[256 + threadIdx.x] += red[256 + threadIdx.x + s];
      red[512 + threadIdx.x] += red[512 + threadIdx.x + s];
    }
    __syncthreads();
  }
  if (threadIdx.x == 0) {
    joints[j * 3 + 0] = red[0];
    joints[j * 3 + 1] = red[256];
    joints[j * 3 + 2] = red[512];
  }
}

// =====================================================================
// 2) xyz0 = v_template + shapedirs @ beta   (flat over N*3)
// =====================================================================
__global__ __launch_bounds__(256) void k_shape(const float* __restrict__ vt,
                                               const float* __restrict__ sd,
                                               const float* __restrict__ beta,
                                               float* __restrict__ xyz0) {
  int i = blockIdx.x * 256 + threadIdx.x;
  if (i >= NV * 3) return;
  float a = vt[i];
  const float* s = &sd[(size_t)i * 10];
#pragma unroll
  for (int l = 0; l < 10; ++l) a += s[l] * beta[l];
  xyz0[i] = a;
}

// =====================================================================
// 3) per-frame pose: rodrigues, pose_feature, kinematic chain -> A
// =====================================================================
__global__ __launch_bounds__(32) void k_pose(const float* __restrict__ body_pose,
                                             const int* __restrict__ parents,
                                             const float* __restrict__ joints,
                                             float* __restrict__ pf,     // [8][207]
                                             float* __restrict__ Amat) { // [8][24][16]
  __shared__ float rm[NJ][9];
  __shared__ float Tw[NJ][16];
  const int t = blockIdx.x;
  const int j = threadIdx.x;
  if (j < NJ) {
    float rx = body_pose[t * 72 + j * 3 + 0];
    float ry = body_pose[t * 72 + j * 3 + 1];
    float rz = body_pose[t * 72 + j * 3 + 2];
    float ang = sqrtf(rx * rx + ry * ry + rz * rz) + 1e-8f;
    float ax = rx / ang, ay = ry / ang, az = rz / ang;
    float s = sinf(ang), c = cosf(ang), cc = 1.f - c;
    float R[9];
    R[0] = 1.f + cc * (-(ay * ay + az * az));
    R[1] = -s * az + cc * (ax * ay);
    R[2] =  s * ay + cc * (ax * az);
    R[3] =  s * az + cc * (ax * ay);
    R[4] = 1.f + cc * (-(ax * ax + az * az));
    R[5] = -s * ax + cc * (ay * az);
    R[6] = -s * ay + cc * (ax * az);
    R[7] =  s * ax + cc * (ay * az);
    R[8] = 1.f + cc * (-(ax * ax + ay * ay));
#pragma unroll
    for (int e = 0; e < 9; ++e) rm[j][e] = R[e];
    if (j >= 1) {
#pragma unroll
      for (int e = 0; e < 9; ++e)
        pf[t * 207 + (j - 1) * 9 + e] =
            R[e] - ((e == 0 || e == 4 || e == 8) ? 1.f : 0.f);
    }
  }
  __syncthreads();
  if (j == 0) {
    for (int q = 0; q < NJ; ++q) {
      int p = parents[q];
      float relx, rely, relz;
      if (q == 0) {
        relx = joints[0]; rely = joints[1]; relz = joints[2];
      } else {
        relx = joints[q * 3 + 0] - joints[p * 3 + 0];
        rely = joints[q * 3 + 1] - joints[p * 3 + 1];
        relz = joints[q * 3 + 2] - joints[p * 3 + 2];
      }
      float tm[16] = {rm[q][0], rm[q][1], rm[q][2], relx,
                      rm[q][3], rm[q][4], rm[q][5], rely,
                      rm[q][6], rm[q][7], rm[q][8], relz,
                      0.f, 0.f, 0.f, 1.f};
      if (q == 0) {
        for (int e = 0; e < 16; ++e) Tw[0][e] = tm[e];
      } else {
        for (int r = 0; r < 4; ++r)
          for (int cl = 0; cl < 4; ++cl) {
            float acc = 0.f;
            for (int m = 0; m < 4; ++m) acc += Tw[p][r * 4 + m] * tm[m * 4 + cl];
            Tw[q][r * 4 + cl] = acc;
          }
      }
    }
    for (int q = 0; q < NJ; ++q) {
      float jx = joints[q * 3 + 0], jy = joints[q * 3 + 1], jz = joints[q * 3 + 2];
      float A[16];
      for (int e = 0; e < 16; ++e) A[e] = Tw[q][e];
      A[3]  -= A[0] * jx + A[1] * jy + A[2]  * jz;
      A[7]  -= A[4] * jx + A[5] * jy + A[6]  * jz;
      A[11] -= A[8] * jx + A[9] * jy + A[10] * jz;
      for (int e = 0; e < 16; ++e) Amat[(t * NJ + q) * 16 + e] = A[e];
    }
  }
}

// =====================================================================
// 4) W1 f32 -> f16, K padded to KP (same row-major [k][n] layout)
// =====================================================================
__global__ __launch_bounds__(256) void k_w1cast(const float* __restrict__ W1,
                                                _Float16* __restrict__ w1h) {
  int i = blockIdx.x * 256 + threadIdx.x;
  if (i >= KP * HID) return;
  int k = i / HID;
  w1h[i] = (k < INDIM) ? (_Float16)W1[i] : (_Float16)0.f;
}

// =====================================================================
// 5) per-vertex LBS + projection + bilinear sampling -> inp (K-major f16)
//    inp layout: [KP][NP] so writes are coalesced across vertices.
// =====================================================================
__device__ __forceinline__ void sample_level(const float* __restrict__ f, int C,
                                             int H, int W, float u, float v,
                                             _Float16* __restrict__ inp,
                                             int col0, int n) {
  float x = fminf(fmaxf(u, 0.f), (float)(W - 1));
  float y = fminf(fmaxf(v, 0.f), (float)(H - 1));
  float xf = floorf(x), yf = floorf(y);
  float wx = x - xf, wy = y - yf;
  int x0 = (int)xf, y0 = (int)yf;
  int x1 = x0 + 1; if (x1 > W - 1) x1 = W - 1;
  int y1 = y0 + 1; if (y1 > H - 1) y1 = H - 1;
  float w00 = (1.f - wx) * (1.f - wy);
  float w01 = wx * (1.f - wy);
  float w10 = (1.f - wx) * wy;
  float w11 = wx * wy;
  size_t i00 = (size_t)y0 * W + x0, i01 = (size_t)y0 * W + x1;
  size_t i10 = (size_t)y1 * W + x0, i11 = (size_t)y1 * W + x1;
  for (int c = 0; c < C; ++c) {
    const float* fc = f + (size_t)c * H * W;
    float val = fc[i00] * w00 + fc[i01] * w01 + fc[i10] * w10 + fc[i11] * w11;
    inp[(size_t)(col0 + c) * NP + n] = (_Float16)val;
  }
}

__global__ __launch_bounds__(128) void k_sample(
    const float* __restrict__ trans, const float* __restrict__ Kmat,
    const float* __restrict__ Emat, const float* __restrict__ posedirs,
    const float* __restrict__ lbsw, const float* __restrict__ feat1,
    const float* __restrict__ feat2, const float* __restrict__ feat3,
    const float* __restrict__ xyz0, const float* __restrict__ pf,
    const float* __restrict__ Amat, _Float16* __restrict__ inp) {
  __shared__ float sPF[TCLIP * 207];
  __shared__ float sA[TCLIP * NJ * 16];
  for (int i = threadIdx.x; i < TCLIP * 207; i += 128) sPF[i] = pf[i];
  for (int i = threadIdx.x; i < TCLIP * NJ * 16; i += 128) sA[i] = Amat[i];
  __syncthreads();

  const int n = blockIdx.x * 128 + threadIdx.x;
  if (n >= NP) return;
  if (n >= NV) {  // zero pad rows so GEMM sees finite data
    for (int k = 0; k < KP; ++k) inp[(size_t)k * NP + n] = (_Float16)0.f;
    return;
  }

  const float px = xyz0[n * 3 + 0], py = xyz0[n * 3 + 1], pz = xyz0[n * 3 + 2];

  // pose offsets for all 8 frames, streaming posedirs exactly once
  float po[TCLIP][3];
#pragma unroll
  for (int t = 0; t < TCLIP; ++t) { po[t][0] = 0.f; po[t][1] = 0.f; po[t][2] = 0.f; }
  for (int k = 0; k < 207; ++k) {
    const float* pd = &posedirs[(size_t)k * (NV * 3) + (size_t)n * 3];
    float d0 = pd[0], d1 = pd[1], d2 = pd[2];
#pragma unroll
    for (int t = 0; t < TCLIP; ++t) {
      float w = sPF[t * 207 + k];
      po[t][0] += w * d0; po[t][1] += w * d1; po[t][2] += w * d2;
    }
  }

  float lw[NJ];
#pragma unroll
  for (int j = 0; j < NJ; ++j) lw[j] = lbsw[(size_t)n * NJ + j];

  for (int t = 0; t < TCLIP; ++t) {
    float T[12];
#pragma unroll
    for (int e = 0; e < 12; ++e) T[e] = 0.f;
    for (int j = 0; j < NJ; ++j) {
      float w = lw[j];
      const float* Aj = &sA[(t * NJ + j) * 16];
#pragma unroll
      for (int e = 0; e < 12; ++e) T[e] += w * Aj[e];
    }
    float qx = px + po[t][0], qy = py + po[t][1], qz = pz + po[t][2];
    float X = T[0] * qx + T[1] * qy + T[2]  * qz + T[3]  + trans[t * 3 + 0];
    float Y = T[4] * qx + T[5] * qy + T[6]  * qz + T[7]  + trans[t * 3 + 1];
    float Z = T[8] * qx + T[9] * qy + T[10] * qz + T[11] + trans[t * 3 + 2];
    const float* E = &Emat[t * 16];
    float cx = E[0] * X + E[1] * Y + E[2]  * Z + E[3];
    float cy = E[4] * X + E[5] * Y + E[6]  * Z + E[7];
    float cz = E[8] * X + E[9] * Y + E[10] * Z + E[11];
    const float* Kc = &Kmat[t * 9];
    float u0 = Kc[0] * cx + Kc[1] * cy + Kc[2] * cz;
    float v0 = Kc[3] * cx + Kc[4] * cy + Kc[5] * cz;
    float h0 = Kc[6] * cx + Kc[7] * cy + Kc[8] * cz;
    float inv = 1.f / (h0 + 1e-8f);
    float u = u0 * inv, v = v0 * inv;
    const int colbase = t * 291;
    inp[(size_t)(colbase + 288) * NP + n] = (_Float16)X;
    inp[(size_t)(colbase + 289) * NP + n] = (_Float16)Y;
    inp[(size_t)(colbase + 290) * NP + n] = (_Float16)Z;
    sample_level(feat1 + (size_t)t * 64 * 256 * 256,  64, 256, 256,
                 u * 0.5f,   v * 0.5f,   inp, colbase + 0,   n);
    sample_level(feat2 + (size_t)t * 96 * 128 * 128,  96, 128, 128,
                 u * 0.25f,  v * 0.25f,  inp, colbase + 64,  n);
    sample_level(feat3 + (size_t)t * 128 * 64 * 64,  128,  64,  64,
                 u * 0.125f, v * 0.125f, inp, colbase + 160, n);
  }

  // static tail: xyz0, rot0=(1,0,0,0), color0, scale0, opacity0, K-padding
  inp[(size_t)2328 * NP + n] = (_Float16)px;
  inp[(size_t)2329 * NP + n] = (_Float16)py;
  inp[(size_t)2330 * NP + n] = (_Float16)pz;
  inp[(size_t)2331 * NP + n] = (_Float16)1.f;
#pragma unroll
  for (int k = 2332; k <= 2340; ++k) inp[(size_t)k * NP + n] = (_Float16)0.f;
  inp[(size_t)2341 * NP + n] = (_Float16)(-2.1972245773362196f);  // log(0.1/0.9)
#pragma unroll
  for (int k = INDIM; k < KP; ++k) inp[(size_t)k * NP + n] = (_Float16)0.f;
}

// =====================================================================
// 6) WMMA GEMM: h = relu(inp^T(K-major) @ W1h + b1), f16 in, f32 acc
//    Block tile 128x64, 8 waves (4x2), each wave 32x32 = 4 C tiles.
//    Double-buffered LDS, register-staged global loads, 1 barrier/K-step.
// =====================================================================
__global__ __launch_bounds__(256) void k_gemm1(const _Float16* __restrict__ inp,  // [KP][NP]
                                               const _Float16* __restrict__ w1h,  // [KP][HID]
                                               const float* __restrict__ b1,
                                               _Float16* __restrict__ h) {        // [NP][HID]
  // LDS tiles stored M-major / N-major so a lane's f16 (k,k+1) pair is one dword
  __shared__ __align__(16) _Float16 sA[2][128 * 34];  // [m][k], stride 34 halfs
  __shared__ __align__(16) _Float16 sB[2][64 * 34];   // [n][k]
  const int tid  = threadIdx.x;
  const int wv   = tid >> 5, lane = tid & 31;
  const int wm   = wv & 3,  wn   = wv >> 2;
  const int hlf  = lane >> 4, l16 = lane & 15;
  const int blockM = blockIdx.x * 128;
  const int blockN = blockIdx.y * 64;

  const unsigned* inpU = (const unsigned*)inp;
  const unsigned* w1U  = (const unsigned*)w1h;

  v8f c00 = {}, c01 = {}, c10 = {}, c11 = {};

  // A tile: 32 k-rows x 64 dwords(128 halfs) = 2048 dwords -> 2 uint4/thread
  const int ka0 = tid >> 4;                    // k row for first uint4
  const int ca0 = (tid & 15) * 4;              // dword col
  const int ka1 = (tid + 256) >> 4;
  const int ca1 = ((tid + 256) & 15) * 4;
  // B tile: 32 k-rows x 32 dwords = 1024 dwords -> 1 uint4/thread
  const int kbr = tid >> 3;
  const int cb  = (tid & 7) * 4;

  uint4 qa0, qa1, qb;

  auto ldg = [&](int kb) {
    qa0 = *(const uint4*)&inpU[(size_t)(kb * 32 + ka0) * (NP / 2) + (blockM >> 1) + ca0];
    qa1 = *(const uint4*)&inpU[(size_t)(kb * 32 + ka1) * (NP / 2) + (blockM >> 1) + ca1];
    qb  = *(const uint4*)&w1U [(size_t)(kb * 32 + kbr) * (HID / 2) + (blockN >> 1) + cb];
  };
  auto sts = [&](int buf) {
    union { uint4 q; _Float16 hh[8]; } u;
    u.q = qa0;
#pragma unroll
    for (int e = 0; e < 8; ++e) sA[buf][(ca0 * 2 + e) * 34 + ka0] = u.hh[e];
    u.q = qa1;
#pragma unroll
    for (int e = 0; e < 8; ++e) sA[buf][(ca1 * 2 + e) * 34 + ka1] = u.hh[e];
    u.q = qb;
#pragma unroll
    for (int e = 0; e < 8; ++e) sB[buf][(cb * 2 + e) * 34 + kbr] = u.hh[e];
  };

  ldg(0);
  sts(0);
  const int NKB = KP / 32;  // 74
  for (int kb = 0; kb < NKB; ++kb) {
    const int cur = kb & 1, nxt = cur ^ 1;
    __syncthreads();  // buf[cur] visible to all waves
    if (kb + 1 < NKB) {
      ldg(kb + 1);  // global loads overlap the WMMAs below
      __builtin_prefetch(&inpU[(size_t)((kb + 2) * 32 + ka0) * (NP / 2) + (blockM >> 1) + ca0], 0, 3);
    }

    // ---- fragments (ISA 7.12.2 wave32 layouts) ----
    union { v16h v; unsigned u[8]; } fa0, fa1, fb0, fb1;
    const unsigned* sAu = (const unsigned*)sA[cur];
    const unsigned* sBu = (const unsigned*)sB[cur];
    const int m0 = wm * 32 + l16, m1 = m0 + 16;
    const int n0 = wn * 32 + l16, n1 = n0 + 16;
#pragma unroll
    for (int v = 0; v < 8; ++v) {
      int kk = (v < 4) ? (hlf * 8 + 2 * v) : (16 + hlf * 8 + 2 * (v - 4));
      fa0.u[v] = sAu[(m0 * 34 + kk) >> 1];
      fa1.u[v] = sAu[(m1 * 34 + kk) >> 1];
    }
#pragma unroll
    for (int v = 0; v < 8; ++v) {
      int kk = (v < 4) ? (hlf * 8 + 2 * v) : (16 + hlf * 8 + 2 * (v - 4));
      fb0.u[v] = sBu[(n0 * 34 + kk) >> 1];
      fb1.u[v] = sBu[(n1 * 34 + kk) >> 1];
    }
    c00 = __builtin_amdgcn_wmma_f32_16x16x32_f16(false, fa0.v, false, fb0.v,
                                                 (short)0, c00, false, false);
    c01 = __builtin_amdgcn_wmma_f32_16x16x32_f16(false, fa0.v, false, fb1.v,
                                                 (short)0, c01, false, false);
    c10 = __builtin_amdgcn_wmma_f32_16x16x32_f16(false, fa1.v, false, fb0.v,
                                                 (short)0, c10, false, false);
    c11 = __builtin_amdgcn_wmma_f32_16x16x32_f16(false, fa1.v, false, fb1.v,
                                                 (short)0, c11, false, false);
    if (kb + 1 < NKB) sts(nxt);  // fill the other buffer; no barrier needed here
  }

  // ---- epilogue: bias + ReLU, store f16 ----
  const int nc0 = blockN + wn * 32 + l16;
  const int nc1 = nc0 + 16;
  const float bb0 = b1[nc0];
  const float bb1 = b1[nc1];
#pragma unroll
  for (int rr = 0; rr < 8; ++rr) {
    const int mr0 = blockM + wm * 32 + hlf * 8 + rr;
    const int mr1 = mr0 + 16;
    float v00 = c00[rr] + bb0; v00 = v00 > 0.f ? v00 : 0.f;
    float v01 = c01[rr] + bb1; v01 = v01 > 0.f ? v01 : 0.f;
    float v10 = c10[rr] + bb0; v10 = v10 > 0.f ? v10 : 0.f;
    float v11 = c11[rr] + bb1; v11 = v11 > 0.f ? v11 : 0.f;
    h[(size_t)mr0 * HID + nc0] = (_Float16)v00;
    h[(size_t)mr0 * HID + nc1] = (_Float16)v01;
    h[(size_t)mr1 * HID + nc0] = (_Float16)v10;
    h[(size_t)mr1 * HID + nc1] = (_Float16)v11;
  }
}

// =====================================================================
// 7) layer 2 (512x14) + residual xyz0, f32 output N x 14
// =====================================================================
__global__ __launch_bounds__(128) void k_out(const _Float16* __restrict__ h,
                                             const float* __restrict__ W2,
                                             const float* __restrict__ b2,
                                             const float* __restrict__ xyz0,
                                             float* __restrict__ out) {
  __shared__ float sW2[HID * 14];
  __shared__ __align__(16) _Float16 sH[128 * 64];
  for (int i = threadIdx.x; i < HID * 14; i += 128) sW2[i] = W2[i];
  const int n0 = blockIdx.x * 128;
  const int n  = n0 + threadIdx.x;
  float acc[14];
#pragma unroll
  for (int j = 0; j < 14; ++j) acc[j] = b2[j];

  const unsigned* hu = (const unsigned*)h;
  for (int kb = 0; kb < HID; kb += 64) {
    __syncthreads();
    // stage h[n0..n0+127][kb..kb+63] into LDS (coalesced uint4 loads)
    for (int i = threadIdx.x; i < 128 * 8; i += 128) {
      int row = i >> 3, cq = (i & 7) * 4;
      int rn = n0 + row;
      uint4 q;
      if (rn < NV)
        q = *(const uint4*)&hu[(size_t)rn * (HID / 2) + (kb >> 1) + cq];
      else
        q = uint4{0u, 0u, 0u, 0u};
      *(uint4*)&((unsigned*)sH)[row * 32 + cq] = q;
    }
    __syncthreads();
    if (n < NV) {
      for (int c = 0; c < 64; ++c) {
        float hv = (float)sH[threadIdx.x * 64 + c];
        const float* wr = &sW2[(kb + c) * 14];
#pragma unroll
        for (int j = 0; j < 14; ++j) acc[j] += hv * wr[j];
      }
    }
  }
  if (n < NV) {
    out[(size_t)n * 14 + 0] = xyz0[n * 3 + 0] + acc[0];
    out[(size_t)n * 14 + 1] = xyz0[n * 3 + 1] + acc[1];
    out[(size_t)n * 14 + 2] = xyz0[n * 3 + 2] + acc[2];
#pragma unroll
    for (int j = 3; j < 14; ++j) out[(size_t)n * 14 + j] = acc[j];
  }
}

// =====================================================================
// launcher
// =====================================================================
extern "C" void kernel_launch(void* const* d_in, const int* in_sizes, int n_in,
                              void* d_out, int out_size, void* d_ws, size_t ws_size,
                              hipStream_t stream) {
  const float* body_pose  = (const float*)d_in[0];
  const float* trans      = (const float*)d_in[1];
  const float* beta       = (const float*)d_in[2];
  const float* Kmat       = (const float*)d_in[3];
  const float* Emat       = (const float*)d_in[4];
  const float* v_template = (const float*)d_in[5];
  const float* shapedirs  = (const float*)d_in[6];
  const float* posedirs   = (const float*)d_in[7];
  const float* Jreg       = (const float*)d_in[8];
  const float* lbsw       = (const float*)d_in[9];
  const float* feat1      = (const float*)d_in[10];
  const float* feat2      = (const float*)d_in[11];
  const float* feat3      = (const float*)d_in[12];
  const float* W1         = (const float*)d_in[13];
  const float* b1         = (const float*)d_in[14];
  const float* W2         = (const float*)d_in[15];
  const float* b2         = (const float*)d_in[16];
  const int*   parents    = (const int*)d_in[17];

  char* ws = (char*)d_ws;
  size_t off = 0;
  auto alloc = [&](size_t bytes) {
    size_t ret = off;
    off += (bytes + 255) & ~(size_t)255;
    return ret;
  };
  float*     w_joints = (float*)(ws + alloc(NJ * 3 * sizeof(float)));
  float*     w_pf     = (float*)(ws + alloc(TCLIP * 207 * sizeof(float)));
  float*     w_A      = (float*)(ws + alloc(TCLIP * NJ * 16 * sizeof(float)));
  float*     w_xyz0   = (float*)(ws + alloc((size_t)NV * 3 * sizeof(float)));
  _Float16*  w_W1h    = (_Float16*)(ws + alloc((size_t)KP * HID * sizeof(_Float16)));
  _Float16*  w_inp    = (_Float16*)(ws + alloc((size_t)KP * NP * sizeof(_Float16)));
  _Float16*  w_h      = (_Float16*)(ws + alloc((size_t)NP * HID * sizeof(_Float16)));

  k_joints<<<NJ, 256, 0, stream>>>(Jreg, v_template, w_joints);
  k_shape<<<(NV * 3 + 255) / 256, 256, 0, stream>>>(v_template, shapedirs, beta, w_xyz0);
  k_pose<<<TCLIP, 32, 0, stream>>>(body_pose, parents, w_joints, w_pf, w_A);
  k_w1cast<<<(KP * HID + 255) / 256, 256, 0, stream>>>(W1, w_W1h);
  k_sample<<<(NP + 127) / 128, 128, 0, stream>>>(trans, Kmat, Emat, posedirs, lbsw,
                                                 feat1, feat2, feat3, w_xyz0, w_pf,
                                                 w_A, w_inp);
  dim3 g1(NP / 128, HID / 64);
  k_gemm1<<<g1, 256, 0, stream>>>(w_inp, w_W1h, b1, w_h);
  k_out<<<(NV + 127) / 128, 128, 0, stream>>>(w_h, W2, b2, w_xyz0, (float*)d_out);
}
